// TorchClassificationModel_65549790871658
// MI455X (gfx1250) — matrix operand order repaired and checked by reading, not compiled, not run on hardware
//
#include <hip/hip_runtime.h>
#include <math.h>

// ---------------------------------------------------------------------------
// Mini-transformer classifier for MI455X (gfx1250, wave32, WMMA).
// B=16, L=1024, E=1024, H=16, DH=64.
//
// Pipeline:
//  1) embed gather -> x bf16                           (bandwidth)
//  2) transpose Wq/Wk/Wv -> bf16 Wt[N][K]; rowmean(Wo); zero r
//  3) q/k/v = x @ W + b via v_wmma_f32_16x16x32_bf16, software-pipelined
//     K-loop (next-step fragments prefetched into regs -> partial loadcnt
//     waits instead of s_wait_loadcnt 0 before every WMMA)
//  4) flash attention per (b,h,16-row q tile) wave: online softmax,
//     P via LDS round-trip, V-fragment loads hoisted to overlap softmax,
//     ctx fused with dot(rowmean(Wo)) -> atomicAdd into r[b,l]
//  5) logits[b] = sum_l Wl[l]*(r[b,l]+mean(bo)) + bl; sigmoid
// ---------------------------------------------------------------------------

typedef __attribute__((ext_vector_type(16))) __bf16 v16bf;
typedef __attribute__((ext_vector_type(8)))  float  v8f;

struct FragU { uint4 lo, hi; };
static_assert(sizeof(FragU) == 32, "frag size");

#define BB 16
#define LL 1024
#define EE 1024
#define HH 16
#define DHH 64

#define WMMA_BF16(A, B, C) \
  __builtin_amdgcn_wmma_f32_16x16x32_bf16(false, (A), false, (B), (short)0, (C), false, false)

__device__ __forceinline__ unsigned short f2bf(float f) {
  unsigned u = __builtin_bit_cast(unsigned, f);
  u += 0x7FFFu + ((u >> 16) & 1u);   // round-to-nearest-even
  return (unsigned short)(u >> 16);
}

__device__ __forceinline__ v16bf load_frag(const unsigned short* p0,
                                           const unsigned short* p1) {
  FragU f;
  f.lo = *(const uint4*)p0;
  f.hi = *(const uint4*)p1;
  return __builtin_bit_cast(v16bf, f);
}

__device__ __forceinline__ float hmax16(float v) {
#pragma unroll
  for (int off = 8; off; off >>= 1) v = fmaxf(v, __shfl_xor(v, off, 32));
  return v;
}
__device__ __forceinline__ float hsum16(float v) {
#pragma unroll
  for (int off = 8; off; off >>= 1) v += __shfl_xor(v, off, 32);
  return v;
}

// ---------------------------------------------------------------------------
// 1) Embedding gather + f32 -> bf16 convert. One block per (b,l) row.
// ---------------------------------------------------------------------------
__global__ __launch_bounds__(256)
void embed_kernel(const int* __restrict__ tokens,
                  const float* __restrict__ table,
                  unsigned short* __restrict__ xbf) {
  const int row = blockIdx.x;                 // b*L + l
  const int tok = tokens[row];
  const float4* src = (const float4*)(table + (size_t)tok * EE);
  const int t = threadIdx.x;                  // 256 threads * 4 elems = 1024
  float4 v = src[t];
  unsigned p0 = (unsigned)f2bf(v.x) | ((unsigned)f2bf(v.y) << 16);
  unsigned p1 = (unsigned)f2bf(v.z) | ((unsigned)f2bf(v.w) << 16);
  uint2 pv; pv.x = p0; pv.y = p1;
  *(uint2*)(xbf + (size_t)row * EE + t * 4) = pv;
}

// ---------------------------------------------------------------------------
// 2a) Weight transpose + convert: Wt[n*E + k] = bf16(W[k*E + n])
// ---------------------------------------------------------------------------
__global__ __launch_bounds__(256)
void wtrans_kernel(const float* __restrict__ W, unsigned short* __restrict__ Wt) {
  __shared__ float tile[32][33];
  const int tx = threadIdx.x, ty = threadIdx.y;
  const int x = blockIdx.x * 32 + tx;
  const int y0 = blockIdx.y * 32;
#pragma unroll
  for (int j = 0; j < 32; j += 8)
    tile[ty + j][tx] = W[(size_t)(y0 + ty + j) * EE + x];
  __syncthreads();
  const int xo = blockIdx.y * 32 + tx;
  const int yo0 = blockIdx.x * 32;
#pragma unroll
  for (int j = 0; j < 32; j += 8)
    Wt[(size_t)(yo0 + ty + j) * EE + xo] = f2bf(tile[tx][ty + j]);
}

// ---------------------------------------------------------------------------
// 2b) rowmean(Wo) and zero the r accumulator (fresh every launch).
// ---------------------------------------------------------------------------
__global__ __launch_bounds__(256)
void prep_misc_kernel(const float* __restrict__ Wo,
                      float* __restrict__ wom,
                      float* __restrict__ r) {
  const int g = blockIdx.x * blockDim.x + threadIdx.x;
  if (g < BB * LL) r[g] = 0.0f;
  if (g < EE) {
    float s = 0.0f;
    for (int e = 0; e < EE; ++e) s += Wo[(size_t)g * EE + e];
    wom[g] = s * (1.0f / (float)EE);
  }
}

// ---------------------------------------------------------------------------
// 3) GEMM: out = x(16384 x 1024) @ W(1024 x 1024) + bias, bf16 WMMA.
//    One wave computes a 16(M) x 64(N) tile; K loop step 32, software
//    pipelined: k+32 fragments are in flight while k's WMMAs execute.
//    vmode 0: store (b,h,l,d)   vmode 1: store transposed (b,h,d,l)
// ---------------------------------------------------------------------------
__global__ __launch_bounds__(256)
void qkv_gemm_kernel(const unsigned short* __restrict__ x,
                     const unsigned short* __restrict__ wt,
                     const float* __restrict__ bias,
                     unsigned short* __restrict__ out,
                     int vmode) {
  const int tid = threadIdx.x;
  const int lane = tid & 31;
  const int wv = tid >> 5;
  const int w = blockIdx.x * 8 + wv;     // 16384 waves total
  const int nt = w & 15;                 // 16 tiles of 64 columns
  const int mt = w >> 4;                 // 1024 tiles of 16 rows
  const int l16 = lane & 15;
  const int lh = lane >> 4;

  v8f acc0 = {}, acc1 = {}, acc2 = {}, acc3 = {};

  const unsigned short* pa_base = x + (size_t)(mt * 16 + l16) * EE + lh * 16;
  const int n0 = nt * 64;
  const unsigned short* pb0_base = wt + (size_t)(n0 + 0 * 16 + l16) * EE + lh * 16;
  const unsigned short* pb1_base = wt + (size_t)(n0 + 1 * 16 + l16) * EE + lh * 16;
  const unsigned short* pb2_base = wt + (size_t)(n0 + 2 * 16 + l16) * EE + lh * 16;
  const unsigned short* pb3_base = wt + (size_t)(n0 + 3 * 16 + l16) * EE + lh * 16;

  // ---- prologue: stage k = 0 fragments -----------------------------------
  v16bf a  = load_frag(pa_base, pa_base + 8);
  v16bf b0 = load_frag(pb0_base, pb0_base + 8);
  v16bf b1 = load_frag(pb1_base, pb1_base + 8);
  v16bf b2 = load_frag(pb2_base, pb2_base + 8);
  v16bf b3 = load_frag(pb3_base, pb3_base + 8);

  // ---- steady state: issue k+32 loads, then consume k --------------------
  for (int k = 0; k < EE - 32; k += 32) {
    const int kn = k + 32;
    const unsigned short* pa = pa_base + kn;
    __builtin_prefetch((const void*)(pa + 128), 0, 3);  // global_prefetch_b8
    v16bf an  = load_frag(pa, pa + 8);
    v16bf bn0 = load_frag(pb0_base + kn, pb0_base + kn + 8);
    v16bf bn1 = load_frag(pb1_base + kn, pb1_base + kn + 8);
    v16bf bn2 = load_frag(pb2_base + kn, pb2_base + kn + 8);
    v16bf bn3 = load_frag(pb3_base + kn, pb3_base + kn + 8);

    acc0 = WMMA_BF16(a, b0, acc0);      // waits only for the older group
    acc1 = WMMA_BF16(a, b1, acc1);
    acc2 = WMMA_BF16(a, b2, acc2);
    acc3 = WMMA_BF16(a, b3, acc3);

    a = an; b0 = bn0; b1 = bn1; b2 = bn2; b3 = bn3;
  }
  // ---- epilogue step (k = EE-32) -----------------------------------------
  acc0 = WMMA_BF16(a, b0, acc0);
  acc1 = WMMA_BF16(a, b1, acc1);
  acc2 = WMMA_BF16(a, b2, acc2);
  acc3 = WMMA_BF16(a, b3, acc3);

  // Epilogue: bias add, convert to bf16, scatter to head-major layout.
#pragma unroll
  for (int t = 0; t < 4; ++t) {
    const v8f acc = (t == 0) ? acc0 : (t == 1) ? acc1 : (t == 2) ? acc2 : acc3;
    const int col = n0 + t * 16 + l16;
    const int h = col >> 6;
    const int d = col & 63;
    const float bi = bias[col];
#pragma unroll
    for (int j = 0; j < 8; ++j) {
      const int rowg = mt * 16 + j + 8 * lh;
      const int b = rowg >> 10;
      const int l = rowg & 1023;
      const unsigned short v = f2bf(acc[j] + bi);
      if (vmode == 0)
        out[((size_t)(b * HH + h) * LL + l) * DHH + d] = v;
      else
        out[((size_t)(b * HH + h) * DHH + d) * LL + l] = v;
    }
  }
}

// ---------------------------------------------------------------------------
// 4) Flash attention. One wave = one (b, h, 16-row Q tile).
//    Online softmax over 32-key chunks; P via wave-private LDS round trip;
//    V fragments loaded early so their latency hides behind softmax VALU.
//    Context fused against rowmean(Wo) -> atomicAdd into r[b,l].
// ---------------------------------------------------------------------------
__global__ __launch_bounds__(256)
void attn_kernel(const unsigned short* __restrict__ qh,
                 const unsigned short* __restrict__ kh,
                 const unsigned short* __restrict__ vt,
                 const float* __restrict__ wom,
                 float* __restrict__ r) {
  __shared__ unsigned short lds_p[8][16 * 32];   // per-wave 16x32 bf16 P tile
  const int tid = threadIdx.x;
  const int lane = tid & 31;
  const int wv = tid >> 5;
  unsigned short* pbuf = lds_p[wv];

  const int w = blockIdx.x * 8 + wv;     // 16384 waves
  const int qt = w & 63;
  const int h = (w >> 6) & 15;
  const int b = w >> 10;
  const int l16 = lane & 15;
  const int lh = lane >> 4;

  // Q fragments: A-layout, d = 0..31 and 32..63
  const size_t qbase = ((size_t)(b * HH + h) * LL + qt * 16 + l16) * DHH;
  const v16bf aq0 = load_frag(qh + qbase + 0 * 32 + lh * 16,
                              qh + qbase + 0 * 32 + lh * 16 + 8);
  const v16bf aq1 = load_frag(qh + qbase + 1 * 32 + lh * 16,
                              qh + qbase + 1 * 32 + lh * 16 + 8);

  v8f acc0 = {}, acc1 = {}, acc2 = {}, acc3 = {};
  float mrun[8], srun[8];
#pragma unroll
  for (int j = 0; j < 8; ++j) { mrun[j] = -3.0e38f; srun[j] = 0.0f; }

  const size_t khead = (size_t)(b * HH + h) * LL * DHH;   // (l,d) rows
  const size_t vhead = (size_t)(b * HH + h) * DHH * LL;   // (d,l) rows

  for (int j0 = 0; j0 < LL; j0 += 32) {
    // ---- K fragments for 32 keys -----------------------------------------
    const unsigned short* kpA = kh + khead + (size_t)(j0 + 0 + l16) * DHH;
    const unsigned short* kpB = kh + khead + (size_t)(j0 + 16 + l16) * DHH;
    v16bf bkA0 = load_frag(kpA + 0 * 32 + lh * 16, kpA + 0 * 32 + lh * 16 + 8);
    v16bf bkA1 = load_frag(kpA + 1 * 32 + lh * 16, kpA + 1 * 32 + lh * 16 + 8);
    v16bf bkB0 = load_frag(kpB + 0 * 32 + lh * 16, kpB + 0 * 32 + lh * 16 + 8);
    v16bf bkB1 = load_frag(kpB + 1 * 32 + lh * 16, kpB + 1 * 32 + lh * 16 + 8);

    // ---- V fragments issued early: latency hides behind scores+softmax ----
    const unsigned short* vp0 = vt + vhead + (size_t)(0 * 16 + l16) * LL + j0 + lh * 16;
    const unsigned short* vp1 = vt + vhead + (size_t)(1 * 16 + l16) * LL + j0 + lh * 16;
    const unsigned short* vp2 = vt + vhead + (size_t)(2 * 16 + l16) * LL + j0 + lh * 16;
    const unsigned short* vp3 = vt + vhead + (size_t)(3 * 16 + l16) * LL + j0 + lh * 16;
    v16bf bv0 = load_frag(vp0, vp0 + 8);
    v16bf bv1 = load_frag(vp1, vp1 + 8);
    v16bf bv2 = load_frag(vp2, vp2 + 8);
    v16bf bv3 = load_frag(vp3, vp3 + 8);

    // ---- scores: two 16x16 WMMA tiles ------------------------------------
    v8f sA = {}, sB = {};
    sA = WMMA_BF16(aq0, bkA0, sA);
    sA = WMMA_BF16(aq1, bkA1, sA);
    sB = WMMA_BF16(aq0, bkB0, sB);
    sB = WMMA_BF16(aq1, bkB1, sB);

    // ---- online softmax over the 32-key slab -----------------------------
    float fac[8];
#pragma unroll
    for (int j = 0; j < 8; ++j) {
      const float scA = sA[j] * 0.125f;   // 1/sqrt(64)
      const float scB = sB[j] * 0.125f;
      const float rm = hmax16(fmaxf(scA, scB));
      const float mnew = fmaxf(mrun[j], rm);
      const float f = __expf(mrun[j] - mnew);
      const float pA = __expf(scA - mnew);
      const float pB = __expf(scB - mnew);
      srun[j] = srun[j] * f + hsum16(pA + pB);
      mrun[j] = mnew;
      fac[j] = f;
      const int prow = j + 8 * lh;
      pbuf[prow * 32 + 0 + l16]  = f2bf(pA);   // ds_store_b16
      pbuf[prow * 32 + 16 + l16] = f2bf(pB);
    }
#pragma unroll
    for (int j = 0; j < 8; ++j) {
      acc0[j] *= fac[j]; acc1[j] *= fac[j]; acc2[j] *= fac[j]; acc3[j] *= fac[j];
    }

    // ---- P(16x32) from LDS in A-layout (same-wave LDS ops are in-order) ---
    const unsigned short* pp = pbuf + l16 * 32 + lh * 16;
    FragU fu;
    fu.lo = *(const uint4*)(pp);        // ds_load_b128
    fu.hi = *(const uint4*)(pp + 8);
    const v16bf ap = __builtin_bit_cast(v16bf, fu);

    // ---- ctx += P @ V over this 32-key slab ------------------------------
    acc0 = WMMA_BF16(ap, bv0, acc0);
    acc1 = WMMA_BF16(ap, bv1, acc1);
    acc2 = WMMA_BF16(ap, bv2, acc2);
    acc3 = WMMA_BF16(ap, bv3, acc3);
  }

  // ---- finalize: ctx/s dotted with rowmean(Wo) for this head ---------------
  const float w0 = wom[h * DHH + 0 * 16 + l16];
  const float w1 = wom[h * DHH + 1 * 16 + l16];
  const float w2 = wom[h * DHH + 2 * 16 + l16];
  const float w3 = wom[h * DHH + 3 * 16 + l16];
#pragma unroll
  for (int j = 0; j < 8; ++j) {
    const float inv = 1.0f / srun[j];
    float part = (acc0[j] * w0 + acc1[j] * w1 + acc2[j] * w2 + acc3[j] * w3) * inv;
    part = hsum16(part);
    if (l16 == 0)
      atomicAdd(&r[b * LL + qt * 16 + j + 8 * lh], part);  // sum over heads
  }
}

// ---------------------------------------------------------------------------
// 5) logits[b] = sum_l Wl[l]*(r[b,l] + mean(bo)) + bl; sigmoid.
// ---------------------------------------------------------------------------
__global__ __launch_bounds__(256)
void final_kernel(const float* __restrict__ r,
                  const float* __restrict__ Wl,
                  const float* __restrict__ bl,
                  const float* __restrict__ bo,
                  float* __restrict__ out) {
  __shared__ float red[256];
  const int t = threadIdx.x;
  float s = 0.0f;
  for (int i = t; i < EE; i += 256) s += bo[i];
  red[t] = s; __syncthreads();
  for (int o = 128; o; o >>= 1) { if (t < o) red[t] += red[t + o]; __syncthreads(); }
  const float mbo = red[0] * (1.0f / (float)EE);
  __syncthreads();

  for (int b = 0; b < BB; ++b) {
    float p = 0.0f;
    for (int l = t; l < LL; l += 256) p += Wl[l] * (r[b * LL + l] + mbo);
    red[t] = p; __syncthreads();
    for (int o = 128; o; o >>= 1) { if (t < o) red[t] += red[t + o]; __syncthreads(); }
    if (t == 0) out[b] = 1.0f / (1.0f + __expf(-(red[0] + bl[0])));
    __syncthreads();
  }
}

// ---------------------------------------------------------------------------
extern "C" void kernel_launch(void* const* d_in, const int* in_sizes, int n_in,
                              void* d_out, int out_size, void* d_ws, size_t ws_size,
                              hipStream_t stream) {
  (void)in_sizes; (void)n_in; (void)out_size; (void)ws_size;

  const int*   tokens = (const int*)d_in[0];
  const float* table  = (const float*)d_in[1];
  const float* Wq = (const float*)d_in[2];
  const float* bq = (const float*)d_in[3];
  const float* Wk = (const float*)d_in[4];
  const float* bk = (const float*)d_in[5];
  const float* Wv = (const float*)d_in[6];
  const float* bv = (const float*)d_in[7];
  const float* Wo = (const float*)d_in[8];
  const float* bo = (const float*)d_in[9];
  const float* Wl = (const float*)d_in[10];
  const float* bl = (const float*)d_in[11];
  float* out = (float*)d_out;

  char* ws = (char*)d_ws;
  size_t off = 0;
  unsigned short* xbf = (unsigned short*)(ws + off); off += (size_t)BB * LL * EE * 2;
  unsigned short* wqT = (unsigned short*)(ws + off); off += (size_t)EE * EE * 2;
  unsigned short* wkT = (unsigned short*)(ws + off); off += (size_t)EE * EE * 2;
  unsigned short* wvT = (unsigned short*)(ws + off); off += (size_t)EE * EE * 2;
  unsigned short* qhm = (unsigned short*)(ws + off); off += (size_t)BB * LL * EE * 2;
  unsigned short* khm = (unsigned short*)(ws + off); off += (size_t)BB * LL * EE * 2;
  unsigned short* vtm = (unsigned short*)(ws + off); off += (size_t)BB * LL * EE * 2;
  float* wom = (float*)(ws + off); off += (size_t)EE * 4;
  float* r   = (float*)(ws + off); off += (size_t)BB * LL * 4;

  // 1) embedding gather -> bf16
  embed_kernel<<<BB * LL, 256, 0, stream>>>(tokens, table, xbf);

  // 2) weight prep
  dim3 tb(32, 8), tg(EE / 32, EE / 32);
  wtrans_kernel<<<tg, tb, 0, stream>>>(Wq, wqT);
  wtrans_kernel<<<tg, tb, 0, stream>>>(Wk, wkT);
  wtrans_kernel<<<tg, tb, 0, stream>>>(Wv, wvT);
  prep_misc_kernel<<<64, 256, 0, stream>>>(Wo, wom, r);

  // 3) q/k/v projections (WMMA bf16, software pipelined)
  qkv_gemm_kernel<<<2048, 256, 0, stream>>>(xbf, wqT, bq, qhm, 0);
  qkv_gemm_kernel<<<2048, 256, 0, stream>>>(xbf, wkT, bk, khm, 0);
  qkv_gemm_kernel<<<2048, 256, 0, stream>>>(xbf, wvT, bv, vtm, 1);

  // 4) flash attention fused with pooled O-projection
  attn_kernel<<<2048, 256, 0, stream>>>(qhm, khm, vtm, wom, r);

  // 5) logits + sigmoid
  final_kernel<<<1, 256, 0, stream>>>(r, Wl, bl, bo, out);
}
